// FixedSAE_46059229282445
// MI455X (gfx1250) — matrix operand order repaired and checked by reading, not compiled, not run on hardware
//
#include <hip/hip_runtime.h>
#include <hip/hip_bf16.h>
#include <stdint.h>

// ---------------------------------------------------------------------------
// TopK SAE forward for MI455X (gfx1250, wave32, WMMA).
//   encode : h = relu(x @ W_enc^T + b_enc)        [65536,1024]
//   topk   : keep top-20 per row, zero rest
//   decode : recon = h @ W_dec^T + b_dec          [65536,768]
// GEMMs: bf16 WMMA (v_wmma_f32_16x16x32_bf16), f32 accumulate, double-buffered
// LDS pipeline; W tiles staged via gfx1250 async global->LDS when available.
// ---------------------------------------------------------------------------

typedef __attribute__((ext_vector_type(16))) __bf16 v16bf;
typedef __attribute__((ext_vector_type(8)))  __bf16 bf16x8;
typedef __attribute__((ext_vector_type(4)))  __bf16 bf16x4;
typedef __attribute__((ext_vector_type(8)))  float  v8f;
typedef int sae_b128 __attribute__((vector_size(16)));   // matches builtin param

#define SAE_B     65536
#define SAE_IN    768
#define SAE_HID   1024
#define SAE_K     20

#if defined(__gfx1250__) &&                                        \
    __has_builtin(__builtin_amdgcn_global_load_async_to_lds_b128) && \
    __has_builtin(__builtin_amdgcn_s_wait_asynccnt)
#define SAE_ASYNC_LDS 1
#else
#define SAE_ASYNC_LDS 0
#endif

// ---------------------------------------------------------------------------
// f32 -> bf16 weight conversion (once per launch; ~3 MB total)
// ---------------------------------------------------------------------------
__global__ __launch_bounds__(256) void sae_cvt_bf16(const float* __restrict__ in,
                                                    __bf16* __restrict__ out, int n4) {
  int i = blockIdx.x * blockDim.x + threadIdx.x;
  if (i < n4) {
    const float4 f = ((const float4*)in)[i];
    bf16x4 o;
    o[0] = (__bf16)f.x; o[1] = (__bf16)f.y; o[2] = (__bf16)f.z; o[3] = (__bf16)f.w;
    *(bf16x4*)(out + (size_t)i * 4) = o;
  }
}

// ---------------------------------------------------------------------------
// Tiled bf16 WMMA GEMM:  C[M,N] = relu?( A[M,K](f32) @ W[N,K](bf16)^T + bias )
// Workgroup tile 128x128, 8 waves each owning 32(M)x64(N) = 2x4 WMMA tiles.
// K in chunks of 64 through double-buffered LDS.
// ---------------------------------------------------------------------------
template <bool RELU>
__global__ __launch_bounds__(256) void sae_gemm_bf16(
    const float*  __restrict__ A,     // [M x K] f32 row-major
    const __bf16* __restrict__ W,     // [N x K] bf16 row-major
    const float*  __restrict__ bias,  // [N]
    float*        __restrict__ C,     // [M x N] f32 row-major
    int M, int N, int K)
{
  __shared__ __attribute__((aligned(16))) __bf16 lds_a[2][128 * 64];
  __shared__ __attribute__((aligned(16))) __bf16 lds_w[2][128 * 64];

  const int tid  = threadIdx.x;
  const int lane = tid & 31;
  const int wv   = tid >> 5;       // 0..7
  const int wm   = wv & 3;         // M sub-tile (32 rows)
  const int wn   = wv >> 2;        // N sub-tile (64 cols)
  const int half = lane >> 4;      // 0: lanes 0-15, 1: lanes 16-31
  const int lrow = lane & 15;

  const int m0 = blockIdx.y * 128;
  const int n0 = blockIdx.x * 128;

  // Per-thread staging coordinates (fixed across chunks).
  const int ar = tid >> 4;               // A: rows ar, ar+16, ... (+16 per iter)
  const int ac = (tid & 15) << 2;        // A: col within chunk (vec4)
  const int wr = tid >> 3;               // W: rows wr, wr+32, ... (+32 per iter)
  const int wc = (tid & 7) << 3;         // W: col within chunk (8 bf16 = 16B)

  float4 areg[8];
#if !SAE_ASYNC_LDS
  uint4 wreg[4];
#endif

  // ---- pipeline helpers ----------------------------------------------------
  auto issueA = [&](int k0) {
#pragma unroll
    for (int it = 0; it < 8; ++it)
      areg[it] = *(const float4*)&A[(size_t)(m0 + ar + it * 16) * K + k0 + ac];
  };
  auto commitA = [&](int buf) {
#pragma unroll
    for (int it = 0; it < 8; ++it) {
      bf16x4 o;
      o[0] = (__bf16)areg[it].x; o[1] = (__bf16)areg[it].y;
      o[2] = (__bf16)areg[it].z; o[3] = (__bf16)areg[it].w;
      *(bf16x4*)&lds_a[buf][(ar + it * 16) * 64 + ac] = o;
    }
  };
  auto issueW = [&](int k0, int buf) {
#if SAE_ASYNC_LDS
#pragma unroll
    for (int it = 0; it < 4; ++it) {
      const __bf16* g = &W[(size_t)(n0 + wr + it * 32) * K + k0 + wc];
      __bf16* l = &lds_w[buf][(wr + it * 32) * 64 + wc];
      __builtin_amdgcn_global_load_async_to_lds_b128(
          (sae_b128*)g, (sae_b128*)l, 0, 0);
    }
#else
#pragma unroll
    for (int it = 0; it < 4; ++it)
      wreg[it] = *(const uint4*)&W[(size_t)(n0 + wr + it * 32) * K + k0 + wc];
#endif
  };
  auto commitW = [&](int buf) {
#if SAE_ASYNC_LDS
    __builtin_amdgcn_s_wait_asynccnt(0);
#else
#pragma unroll
    for (int it = 0; it < 4; ++it)
      *(uint4*)&lds_w[buf][(wr + it * 32) * 64 + wc] = wreg[it];
#endif
  };

  // ---- accumulators --------------------------------------------------------
  v8f c[2][4];
#pragma unroll
  for (int mt = 0; mt < 2; ++mt)
#pragma unroll
    for (int nt = 0; nt < 4; ++nt)
#pragma unroll
      for (int r = 0; r < 8; ++r) c[mt][nt][r] = 0.0f;

  // ---- prologue: fill buffer 0 --------------------------------------------
  issueA(0);
  issueW(0, 0);
  commitA(0);
  commitW(0);
  __syncthreads();

  int buf = 0;
  for (int k0 = 0; k0 < K; k0 += 64) {
    const int  nxt  = buf ^ 1;
    const bool more = (k0 + 64 < K);

    // issue next chunk's global traffic before computing on current chunk
    if (more) { issueA(k0 + 64); issueW(k0 + 64, nxt); }

    const __bf16* la = lds_a[buf];
    const __bf16* lw = lds_w[buf];

#pragma unroll
    for (int s = 0; s < 2; ++s) {
      v16bf a[2], b[4];
      // A fragment (ISA 7.12.2, 16-bit A 16x32): lane holds row lrow,
      // K = 8*half + {0..7} and 8*half + 16 + {0..7}
#pragma unroll
      for (int mt = 0; mt < 2; ++mt) {
        const __bf16* p = &la[(wm * 32 + mt * 16 + lrow) * 64 + s * 32 + half * 8];
        bf16x8 lo = *(const bf16x8*)p;
        bf16x8 hi = *(const bf16x8*)(p + 16);
        a[mt] = __builtin_shufflevector(lo, hi, 0, 1, 2, 3, 4, 5, 6, 7,
                                        8, 9, 10, 11, 12, 13, 14, 15);
      }
      // B fragment: lane holds column lrow, K = 8*half + {0..7} and 8*half + 16 + {0..7}
#pragma unroll
      for (int nt = 0; nt < 4; ++nt) {
        const __bf16* p = &lw[(wn * 64 + nt * 16 + lrow) * 64 + s * 32 + half * 8];
        bf16x8 lo = *(const bf16x8*)p;
        bf16x8 hi = *(const bf16x8*)(p + 16);
        b[nt] = __builtin_shufflevector(lo, hi, 0, 1, 2, 3, 4, 5, 6, 7,
                                        8, 9, 10, 11, 12, 13, 14, 15);
      }
#pragma unroll
      for (int mt = 0; mt < 2; ++mt)
#pragma unroll
        for (int nt = 0; nt < 4; ++nt)
          c[mt][nt] = __builtin_amdgcn_wmma_f32_16x16x32_bf16(
              false, a[mt], false, b[nt], (short)0, c[mt][nt], false, false);
    }

    // commit next chunk to its LDS buffer, then one barrier per iteration
    if (more) { commitA(nxt); commitW(nxt); }
    __syncthreads();
    buf = nxt;
  }

  // ---- epilogue: bias (+relu), f32 store ----------------------------------
#pragma unroll
  for (int nt = 0; nt < 4; ++nt) {
    int col  = n0 + wn * 64 + nt * 16 + lrow;
    float bv = bias[col];
#pragma unroll
    for (int mt = 0; mt < 2; ++mt) {
#pragma unroll
      for (int r = 0; r < 8; ++r) {
        int row = m0 + wm * 32 + mt * 16 + half * 8 + r;   // C layout: M = r + 8*half
        float v = c[mt][nt][r] + bv;
        if (RELU) v = fmaxf(v, 0.0f);
        C[(size_t)row * N + col] = v;
      }
    }
  }
}

// ---------------------------------------------------------------------------
// TopK mask, in place. One wave32 per row of h[65536,1024]. Lane owns 32
// values (col = s*32 + lane). 20 rounds of cross-lane argmax on packed
// (value_bits, 1023-col) keys: earlier index wins ties, like lax.top_k.
// All values are >= 0 (post-ReLU), so float bits compare monotonically.
// ---------------------------------------------------------------------------
__global__ __launch_bounds__(256) void sae_topk_mask(float* __restrict__ h) {
  const int lane = threadIdx.x & 31;
  const int row  = blockIdx.x * 8 + (threadIdx.x >> 5);

  float v[32];
#pragma unroll
  for (int s = 0; s < 32; ++s)
    v[s] = h[(size_t)row * SAE_HID + s * 32 + lane];

  unsigned keep = 0u, dead = 0u;
  for (int it = 0; it < SAE_K; ++it) {
    float bv = -1.0f; int bs = 0;
#pragma unroll
    for (int s = 0; s < 32; ++s) {
      bool alive = !((dead >> s) & 1u);
      if (alive && v[s] > bv) { bv = v[s]; bs = s; }
    }
    unsigned long long key = 0ull;
    if (bv >= 0.0f)
      key = ((unsigned long long)__float_as_uint(bv) << 32) |
            (unsigned)(1023 - (bs * 32 + lane));
    unsigned long long m = key;
#pragma unroll
    for (int off = 16; off >= 1; off >>= 1) {
      unsigned long long o = __shfl_xor(m, off, 32);
      if (o > m) m = o;
    }
    if (m != 0ull && key == m) { keep |= 1u << bs; dead |= 1u << bs; }
  }

#pragma unroll
  for (int s = 0; s < 32; ++s)
    h[(size_t)row * SAE_HID + s * 32 + lane] = ((keep >> s) & 1u) ? v[s] : 0.0f;
}

// ---------------------------------------------------------------------------
// Launch: convert weights -> encoder GEMM (relu) -> topk -> decoder GEMM.
// d_out = [recon (65536*768) | h (65536*1024)], both f32.
// d_ws  = bf16 W_enc (1.5 MB) + bf16 W_dec (1.5 MB).
// ---------------------------------------------------------------------------
extern "C" void kernel_launch(void* const* d_in, const int* in_sizes, int n_in,
                              void* d_out, int out_size, void* d_ws, size_t ws_size,
                              hipStream_t stream) {
  const float* x     = (const float*)d_in[0];
  const float* W_enc = (const float*)d_in[1];
  const float* b_enc = (const float*)d_in[2];
  const float* W_dec = (const float*)d_in[3];
  const float* b_dec = (const float*)d_in[4];

  float* recon = (float*)d_out;
  float* h     = recon + (size_t)SAE_B * SAE_IN;

  __bf16* wenc_bf = (__bf16*)d_ws;
  __bf16* wdec_bf = wenc_bf + (size_t)SAE_HID * SAE_IN;

  const int NW = SAE_HID * SAE_IN;           // 786432
  sae_cvt_bf16<<<NW / 4 / 256, 256, 0, stream>>>(W_enc, wenc_bf, NW / 4);
  sae_cvt_bf16<<<NW / 4 / 256, 256, 0, stream>>>(W_dec, wdec_bf, NW / 4);

  // encode: M=65536, N=1024, K=768, relu
  sae_gemm_bf16<true><<<dim3(SAE_HID / 128, SAE_B / 128), 256, 0, stream>>>(
      x, wenc_bf, b_enc, h, SAE_B, SAE_HID, SAE_IN);

  // topk mask in place on h
  sae_topk_mask<<<SAE_B / 8, 256, 0, stream>>>(h);

  // decode: M=65536, N=768, K=1024, no relu
  sae_gemm_bf16<false><<<dim3(SAE_IN / 128, SAE_B / 128), 256, 0, stream>>>(
      h, wdec_bf, b_dec, recon, SAE_B, SAE_IN, SAE_HID);
}
